// RIM_78907139162601
// MI455X (gfx1250) — matrix operand order repaired and checked by reading, not compiled, not run on hardware
//
#include <hip/hip_runtime.h>
#include <math.h>

typedef _Float16 half_t;
typedef __attribute__((ext_vector_type(16))) _Float16 v16h;
typedef __attribute__((ext_vector_type(8)))  _Float16 v8h;
typedef __attribute__((ext_vector_type(8)))  float    v8f;

// ---------------- problem constants ----------------
constexpr int B_   = 8192;
constexpr int U_   = 6;
constexpr int D_   = 256;
constexpr int H_   = 512;
constexpr int VIN_ = 400;
constexpr int KIN_ = 64;
constexpr int HC_  = 4;
constexpr int KC_  = 32;
constexpr int VC_  = 512;
constexpr int KGATE = 928;   // 416 (padded inputs) + 512 (hs)
constexpr int NQKV  = 2304;  // 128 qc + 128 kc + 2048 vc

// ---------------- workspace layout (bytes) ----------------
constexpr size_t OFF_X16    = 0;
constexpr size_t OFF_HS16   = OFF_X16    + (size_t)B_*D_*2;
constexpr size_t OFF_WKT    = OFF_HS16   + (size_t)B_*U_*H_*2;
constexpr size_t OFF_WVT    = OFF_WKT    + (size_t)128*256*2;
constexpr size_t OFF_WQT    = OFF_WVT    + (size_t)512*256*2;
constexpr size_t OFF_WQKVT  = OFF_WQT    + (size_t)U_*128*512*2;
constexpr size_t OFF_WOT    = OFF_WQKVT  + (size_t)U_*NQKV*512*2;
constexpr size_t OFF_WCAT   = OFF_WOT    + (size_t)U_*512*2048*2;
constexpr size_t OFF_KIN    = OFF_WCAT   + (size_t)U_*2048*KGATE*2;
constexpr size_t OFF_V0     = OFF_KIN    + (size_t)B_*64*4;
constexpr size_t OFF_QB     = OFF_V0     + (size_t)B_*VIN_*4;
constexpr size_t OFF_MASK   = OFF_QB     + (size_t)B_*U_*64*4;
constexpr size_t OFF_HNEW   = OFF_MASK   + (size_t)B_*U_*4;
constexpr size_t OFF_HNEW16 = OFF_HNEW   + (size_t)B_*U_*H_*4;
constexpr size_t OFF_BIG    = OFF_HNEW16 + (size_t)B_*U_*H_*2;
// phase 1 of BIG region: ACT then GATES (both dead after LSTM kernel)
constexpr size_t OFF_ACT    = OFF_BIG;
constexpr size_t OFF_GATES  = OFF_ACT    + (size_t)B_*U_*KGATE*2;
// phase 2 of BIG region: QKV then CTX16
constexpr size_t OFF_QKV    = OFF_BIG;
constexpr size_t OFF_CTX16  = OFF_QKV    + (size_t)B_*U_*NQKV*4;

// ---------------- f32 -> f16 copy ----------------
__global__ void cvt_f16_kernel(const float* __restrict__ src, half_t* __restrict__ dst, long n) {
  long i = (long)blockIdx.x * blockDim.x + threadIdx.x;
  if (i < n) dst[i] = (half_t)src[i];
}

// transpose src[u][k][n] (f32) -> dst[u][n][Kpad] (f16), zero pad k>=K or n>=N
__global__ void transpose_pad_kernel(const float* __restrict__ src, half_t* __restrict__ dst,
                                     int K, int N, int Kpad, int NRows,
                                     long srcStrideU, long dstStrideU) {
  int k = blockIdx.x * 16 + (threadIdx.x & 15);
  int n = blockIdx.y * 16 + (threadIdx.x >> 4);
  int u = blockIdx.z;
  if (k >= Kpad || n >= NRows) return;
  float v = (k < K && n < N) ? src[(size_t)u*srcStrideU + (size_t)k*N + n] : 0.0f;
  dst[(size_t)u*dstStrideU + (size_t)n*Kpad + k] = (half_t)v;
}

// row-major pad-copy: src[u][n][Ksrc] (f32) -> dst[u][n][KdstTotal] segment (f16)
__global__ void copy_pad_kernel(const float* __restrict__ src, half_t* __restrict__ dst,
                                int Ksrc, int Kseg, int KdstTotal,
                                long srcStrideU, long dstStrideU) {
  int k = blockIdx.x * 256 + threadIdx.x;
  int n = blockIdx.y;
  int u = blockIdx.z;
  if (k >= Kseg) return;
  float v = (k < Ksrc) ? src[(size_t)u*srcStrideU + (size_t)n*Ksrc + k] : 0.0f;
  dst[(size_t)u*dstStrideU + (size_t)n*KdstTotal + k] = (half_t)v;
}

// ---------------- WMMA GEMM: C[m][n] = sum_k A[m][k]*Bt[n][k] [+ bias[n]] [+ add[m][n]] -------------
// Block = 256 thr = 8 waves, block tile 128x128, wave tile 64x32 (4x2 WMMA 16x16x32 f16 frags).
// Fragments live in registers; K-loop is ping-pong double buffered so the 12 b128 loads of the
// next K-step are in flight while the current 8 v_wmma execute.
template <bool HAS_BIAS, bool HAS_ADD>
__global__ __launch_bounds__(256)
void wmma_gemm_kernel(const half_t* __restrict__ A, long lda, long sAu,
                      const half_t* __restrict__ Bt, long sBu,
                      const float* __restrict__ bias, long sBiasU,
                      const float* __restrict__ add, long ldAdd, long sAddU,
                      float* __restrict__ C, long ldc, long sCu,
                      int Nreal, int Kpad) {
  const int u = blockIdx.z;
  A  += (size_t)u * sAu;
  Bt += (size_t)u * sBu;
  C  += (size_t)u * sCu;
  const float* biasU = HAS_BIAS ? bias + (size_t)u * sBiasU : nullptr;
  const float* addU  = HAS_ADD  ? add  + (size_t)u * sAddU  : nullptr;

  const int tid  = threadIdx.x;
  const int lane = tid & 31;
  const int wid  = tid >> 5;
  const int wm = wid & 1;
  const int wn = wid >> 1;
  const int m_base = blockIdx.x * 128 + wm * 64;
  const int n_base = blockIdx.y * 128 + wn * 32;

  const int r  = lane & 15;   // row within 16 (A) / col within 16 (B,C)
  const int hi = lane >> 4;   // half-wave select

  // Per-lane base pointers; advance by 32 halfs per K-step.
  const half_t* pa0 = A + (size_t)(m_base +  0 + r) * lda + hi*16;
  const half_t* pa1 = A + (size_t)(m_base + 16 + r) * lda + hi*16;
  const half_t* pa2 = A + (size_t)(m_base + 32 + r) * lda + hi*16;
  const half_t* pa3 = A + (size_t)(m_base + 48 + r) * lda + hi*16;
  const half_t* pb0 = Bt + (size_t)(n_base +  0 + r) * Kpad + hi*16;
  const half_t* pb1 = Bt + (size_t)(n_base + 16 + r) * Kpad + hi*16;

  v8f zero = {};
  v8f acc[4][2];
#pragma unroll
  for (int mi = 0; mi < 4; ++mi)
#pragma unroll
    for (int ni = 0; ni < 2; ++ni) acc[mi][ni] = zero;

  v16h afA[4], bfA[2], afB[4], bfB[2];

#define LOAD_FRAGS(AF, BF, KOFF)                                                   \
  do {                                                                             \
    const half_t* qa0 = pa0 + (KOFF); const half_t* qa1 = pa1 + (KOFF);            \
    const half_t* qa2 = pa2 + (KOFF); const half_t* qa3 = pa3 + (KOFF);            \
    v8h l0 = *(const v8h*)qa0, h0 = *(const v8h*)(qa0 + 8);                        \
    v8h l1 = *(const v8h*)qa1, h1 = *(const v8h*)(qa1 + 8);                        \
    v8h l2 = *(const v8h*)qa2, h2 = *(const v8h*)(qa2 + 8);                        \
    v8h l3 = *(const v8h*)qa3, h3 = *(const v8h*)(qa3 + 8);                        \
    AF[0] = __builtin_shufflevector(l0, h0, 0,1,2,3,4,5,6,7,8,9,10,11,12,13,14,15);\
    AF[1] = __builtin_shufflevector(l1, h1, 0,1,2,3,4,5,6,7,8,9,10,11,12,13,14,15);\
    AF[2] = __builtin_shufflevector(l2, h2, 0,1,2,3,4,5,6,7,8,9,10,11,12,13,14,15);\
    AF[3] = __builtin_shufflevector(l3, h3, 0,1,2,3,4,5,6,7,8,9,10,11,12,13,14,15);\
    BF[0] = *(const v16h*)(pb0 + (KOFF));                                          \
    BF[1] = *(const v16h*)(pb1 + (KOFF));                                          \
    __builtin_prefetch(qa0 + 32, 0, 1);                                            \
    __builtin_prefetch(qa2 + 32, 0, 1);                                            \
    __builtin_prefetch(pb0 + (KOFF) + 32, 0, 1);                                   \
    __builtin_prefetch(pb1 + (KOFF) + 32, 0, 1);                                   \
  } while (0)

#define MMA_STEP(AF, BF)                                                           \
  do {                                                                             \
    _Pragma("unroll")                                                              \
    for (int mi = 0; mi < 4; ++mi)                                                 \
      _Pragma("unroll")                                                            \
      for (int ni = 0; ni < 2; ++ni)                                               \
        acc[mi][ni] = __builtin_amdgcn_wmma_f32_16x16x32_f16(                      \
            false, AF[mi], false, BF[ni], (short)0, acc[mi][ni], false, false);    \
  } while (0)

  const int nk = Kpad >> 5;           // number of 32-wide K steps (>= 8 in all uses)
  LOAD_FRAGS(afA, bfA, 0);
  int s = 0;
  while (true) {
    if (s + 1 < nk) LOAD_FRAGS(afB, bfB, (s + 1) * 32);
    MMA_STEP(afA, bfA);
    ++s; if (s >= nk) break;
    if (s + 1 < nk) LOAD_FRAGS(afA, bfA, (s + 1) * 32);
    MMA_STEP(afB, bfB);
    ++s; if (s >= nk) break;
  }
#undef LOAD_FRAGS
#undef MMA_STEP

  // C layout: VGPR v, lanes 0-15 -> (M=m0+v, N=lane); lanes 16-31 -> (M=m0+8+v, N=lane-16)
  const int nn = lane & 15;
  const int mo = hi * 8;
#pragma unroll
  for (int ni = 0; ni < 2; ++ni) {
    int n = n_base + ni*16 + nn;
    if (n >= Nreal) continue;
    float bb = HAS_BIAS ? biasU[n] : 0.0f;
#pragma unroll
    for (int mi = 0; mi < 4; ++mi) {
#pragma unroll
      for (int v = 0; v < 8; ++v) {
        int m = m_base + mi*16 + mo + v;
        float val = acc[mi][ni][v] + bb;
        if (HAS_ADD) val += addU[(size_t)m * ldAdd + n];
        C[(size_t)m * ldc + n] = val;
      }
    }
  }
}

// ---------------- router: scores, top-k mask, softmax over t, assemble act ----------------
__global__ __launch_bounds__(128)
void router_kernel(const float* __restrict__ kin,   // B x 64
                   const float* __restrict__ qb,    // B x 6 x 64
                   const float* __restrict__ v0,    // B x 400
                   const float* __restrict__ bk,    // 64   (k_in for t=1 slot: x2 row is zero)
                   const float* __restrict__ bv,    // 400  (v_in for t=1 slot)
                   const half_t* __restrict__ hs16, // B x 6 x 512
                   half_t* __restrict__ act,        // B x 6 x 928
                   float* __restrict__ maskg) {     // B x 6
  __shared__ float s_kin[64], s_bk[64];
  __shared__ float s_s0[6], s_s1[6];
  __shared__ float s_p0[6], s_p1[6], s_m[6];
  const int b = blockIdx.x;
  const int t = threadIdx.x;
  if (t < 64) { s_kin[t] = kin[(size_t)b*64 + t]; s_bk[t] = bk[t]; }
  __syncthreads();
  if (t < 6) {
    const float* q = qb + (size_t)b*(U_*64) + t*64;
    float d0 = 0.f, d1 = 0.f;
    for (int j = 0; j < 64; ++j) { float qv = q[j]; d0 += qv*s_kin[j]; d1 += qv*s_bk[j]; }
    s_s0[t] = d0 * (1.0f/64.0f);
    s_s1[t] = d1 * (1.0f/64.0f);
  }
  __syncthreads();
  if (t < 6) {
    // rank of s0[t] among units (top_k tie-break: lower index wins)
    int cnt = 0;
    float su = s_s0[t];
    for (int v = 0; v < 6; ++v) {
      float sv = s_s0[v];
      if (sv > su || (sv == su && v < t)) ++cnt;
    }
    float m = (cnt < 4) ? 1.0f : 0.0f;
    s_m[t] = m;
    maskg[(size_t)b*U_ + t] = m;
    float a = s_s0[t], c = s_s1[t];
    float mx = fmaxf(a, c);
    float e0 = __expf(a - mx), e1 = __expf(c - mx);
    float inv = 1.0f / (e0 + e1);
    s_p0[t] = e0 * inv; s_p1[t] = e1 * inv;
  }
  __syncthreads();
  half_t* arow = act + (size_t)b * (U_*KGATE);
  const half_t* hrow = hs16 + (size_t)b * (U_*H_);
  for (int u = 0; u < U_; ++u) {
    float p0 = s_p0[u], p1 = s_p1[u], m = s_m[u];
    half_t* au = arow + u*KGATE;
    for (int n = t; n < 416; n += 128) {
      float val = 0.0f;
      if (n < VIN_) val = (p0 * v0[(size_t)b*VIN_ + n] + p1 * bv[n]) * m;
      au[n] = (half_t)val;
    }
    for (int n = t; n < H_; n += 128)
      au[416 + n] = hrow[u*H_ + n];
  }
}

// ---------------- LSTM pointwise ----------------
__global__ void lstm_kernel(const float* __restrict__ gates, // B x 6 x 2048
                            const float* __restrict__ cs,    // B x 6 x 512
                            const float* __restrict__ bih,   // 6 x 2048
                            const float* __restrict__ bhh,   // 6 x 2048
                            float* __restrict__ c_out,       // B x 6 x 512 (-> d_out half 2)
                            float* __restrict__ hnew,        // B x 6 x 512
                            half_t* __restrict__ hnew16,
                            long total) {
  long i = (long)blockIdx.x * blockDim.x + threadIdx.x;
  if (i >= total) return;
  int hch = (int)(i % H_);
  int u   = (int)((i / H_) % U_);
  size_t gbase = (size_t)(i / (U_*H_)) * (U_*2048) + (size_t)u * 2048;
  size_t bb = (size_t)u * 2048;
  float gi = gates[gbase + hch]        + bih[bb + hch]        + bhh[bb + hch];
  float gf = gates[gbase + 512 + hch]  + bih[bb + 512 + hch]  + bhh[bb + 512 + hch];
  float gg = gates[gbase + 1024 + hch] + bih[bb + 1024 + hch] + bhh[bb + 1024 + hch];
  float go = gates[gbase + 1536 + hch] + bih[bb + 1536 + hch] + bhh[bb + 1536 + hch];
  float si = 1.0f / (1.0f + __expf(-gi));
  float sf = 1.0f / (1.0f + __expf(-gf));
  float so = 1.0f / (1.0f + __expf(-go));
  float c  = sf * cs[i] + si * tanhf(gg);
  float hn = so * tanhf(c);
  c_out[i]  = c;
  hnew[i]   = hn;
  hnew16[i] = (half_t)hn;
}

// ---------------- 6x6 multi-head attention, one block per (b, head) ----------------
__global__ __launch_bounds__(256)
void attn_kernel(const float* __restrict__ qkv,  // B x 6 x 2304 (qc|kc|vc raw, no bias)
                 const float* __restrict__ bqc,  // 6 x 128
                 const float* __restrict__ bkc,  // 6 x 128
                 const float* __restrict__ bvc,  // 6 x 2048
                 const float* __restrict__ maskg,// B x 6
                 half_t* __restrict__ ctx16) {   // B x 6 x 2048
  const int b = blockIdx.x;
  const int h = blockIdx.y;
  const int t = threadIdx.x;
  __shared__ float s_q[6][32], s_k[6][32];
  __shared__ float s_att[6][6], s_p[6][6];
  __shared__ float s_v[6][512];
  __shared__ float s_m[6];
  const size_t base = (size_t)b * (U_*NQKV);
  if (t < 6) s_m[t] = maskg[(size_t)b*U_ + t];
  if (t < 192) {
    int u = t >> 5, j = t & 31;
    int off = h*KC_ + j;
    s_q[u][j] = qkv[base + (size_t)u*NQKV + off]       + bqc[u*128 + off];
    s_k[u][j] = qkv[base + (size_t)u*NQKV + 128 + off] + bkc[u*128 + off];
  }
  for (int idx = t; idx < U_*VC_; idx += 256) {
    int tt = idx >> 9, d = idx & 511;
    s_v[tt][d] = qkv[base + (size_t)tt*NQKV + 256 + h*VC_ + d] + bvc[tt*(HC_*VC_) + h*VC_ + d];
  }
  __syncthreads();
  if (t < 36) {
    int u = t / 6, tt = t % 6;
    float a = 0.f;
#pragma unroll
    for (int j = 0; j < 32; ++j) a += s_q[u][j] * s_k[tt][j];
    s_att[u][tt] = a * s_m[u] * 0.17677669529663689f;  // qc masked, /sqrt(32)
  }
  __syncthreads();
  if (t < 6) {
    float mx = s_att[t][0];
    for (int tt = 1; tt < 6; ++tt) mx = fmaxf(mx, s_att[t][tt]);
    float e[6], sum = 0.f;
    for (int tt = 0; tt < 6; ++tt) { e[tt] = __expf(s_att[t][tt] - mx); sum += e[tt]; }
    float inv = s_m[t] / sum;   // p = softmax * mask[u]
    for (int tt = 0; tt < 6; ++tt) s_p[t][tt] = e[tt] * inv;
  }
  __syncthreads();
  for (int idx = t; idx < U_*VC_; idx += 256) {
    int u = idx >> 9, d = idx & 511;
    float a = 0.f;
#pragma unroll
    for (int tt = 0; tt < 6; ++tt) a += s_p[u][tt] * s_v[tt][d];
    ctx16[(size_t)b*(U_*HC_*VC_) + (size_t)u*(HC_*VC_) + h*VC_ + d] = (half_t)a;
  }
}

// ---------------- launch ----------------
extern "C" void kernel_launch(void* const* d_in, const int* in_sizes, int n_in,
                              void* d_out, int out_size, void* d_ws, size_t ws_size,
                              hipStream_t stream) {
  (void)in_sizes; (void)n_in; (void)out_size; (void)ws_size;
  const float* x   = (const float*)d_in[2];
  const float* hs  = (const float*)d_in[3];
  const float* cs  = (const float*)d_in[4];
  const float* Wk  = (const float*)d_in[5];
  const float* bk  = (const float*)d_in[6];
  const float* Wv  = (const float*)d_in[7];
  const float* bv  = (const float*)d_in[8];
  const float* Wq  = (const float*)d_in[9];
  const float* bq  = (const float*)d_in[10];
  const float* Wih = (const float*)d_in[11];
  const float* Whh = (const float*)d_in[12];
  const float* bih = (const float*)d_in[13];
  const float* bhh = (const float*)d_in[14];
  const float* Wqc = (const float*)d_in[15];
  const float* bqc = (const float*)d_in[16];
  const float* Wkc = (const float*)d_in[17];
  const float* bkc = (const float*)d_in[18];
  const float* Wvc = (const float*)d_in[19];
  const float* bvc = (const float*)d_in[20];
  const float* Wo  = (const float*)d_in[21];
  const float* bo  = (const float*)d_in[22];

  char* ws = (char*)d_ws;
  half_t* X16    = (half_t*)(ws + OFF_X16);
  half_t* HS16   = (half_t*)(ws + OFF_HS16);
  half_t* WKT    = (half_t*)(ws + OFF_WKT);
  half_t* WVT    = (half_t*)(ws + OFF_WVT);
  half_t* WQT    = (half_t*)(ws + OFF_WQT);
  half_t* WQKVT  = (half_t*)(ws + OFF_WQKVT);
  half_t* WOT    = (half_t*)(ws + OFF_WOT);
  half_t* WCAT   = (half_t*)(ws + OFF_WCAT);
  float*  KINb   = (float*)(ws + OFF_KIN);
  float*  V0b    = (float*)(ws + OFF_V0);
  float*  QBb    = (float*)(ws + OFF_QB);
  float*  MASKb  = (float*)(ws + OFF_MASK);
  float*  HNEWb  = (float*)(ws + OFF_HNEW);
  half_t* HNEW16 = (half_t*)(ws + OFF_HNEW16);
  half_t* ACTb   = (half_t*)(ws + OFF_ACT);
  float*  GATESb = (float*)(ws + OFF_GATES);
  float*  QKVb   = (float*)(ws + OFF_QKV);
  half_t* CTX16b = (half_t*)(ws + OFF_CTX16);

  float* hsfin = (float*)d_out;                       // B x 6 x 512
  float* cnew  = (float*)d_out + (size_t)B_*U_*H_;    // B x 6 x 512

  // 1) activations to f16
  cvt_f16_kernel<<<(B_*D_ + 255)/256, 256, 0, stream>>>(x, X16, (long)B_*D_);
  cvt_f16_kernel<<<(B_*U_*H_ + 255)/256, 256, 0, stream>>>(hs, HS16, (long)B_*U_*H_);

  // 2) weight prep (transpose to Bt[n][Kpad] f16, zero-padded)
  transpose_pad_kernel<<<dim3(16, 8, 1),   256, 0, stream>>>(Wk,  WKT,  256, 64,   256, 128, 0, 0);
  transpose_pad_kernel<<<dim3(16, 32, 1),  256, 0, stream>>>(Wv,  WVT,  256, 400,  256, 512, 0, 0);
  transpose_pad_kernel<<<dim3(32, 8, U_),  256, 0, stream>>>(Wq,  WQT,  512, 64,   512, 128, (long)512*64, (long)128*512);
  transpose_pad_kernel<<<dim3(32, 8, U_),  256, 0, stream>>>(Wqc, WQKVT,              512, 128,  512, 128,  (long)512*128,  (long)NQKV*512);
  transpose_pad_kernel<<<dim3(32, 8, U_),  256, 0, stream>>>(Wkc, WQKVT + 128*512,    512, 128,  512, 128,  (long)512*128,  (long)NQKV*512);
  transpose_pad_kernel<<<dim3(32, 128, U_),256, 0, stream>>>(Wvc, WQKVT + 256*512,    512, 2048, 512, 2048, (long)512*2048, (long)NQKV*512);
  transpose_pad_kernel<<<dim3(128, 32, U_),256, 0, stream>>>(Wo,  WOT, 2048, 512, 2048, 512, (long)2048*512, (long)512*2048);
  copy_pad_kernel<<<dim3(2, 2048, U_), 256, 0, stream>>>(Wih, WCAT,       400, 416, KGATE, (long)2048*400, (long)2048*KGATE);
  copy_pad_kernel<<<dim3(2, 2048, U_), 256, 0, stream>>>(Whh, WCAT + 416, 512, 512, KGATE, (long)2048*512, (long)2048*KGATE);

  // 3) front GEMMs: k_in, v_in(t=0), q_in
  wmma_gemm_kernel<true,false><<<dim3(64, 1, 1), 256, 0, stream>>>(X16, 256, 0, WKT, 0, bk, 0,
      nullptr, 0, 0, KINb, 64, 0, 64, 256);
  wmma_gemm_kernel<true,false><<<dim3(64, 4, 1), 256, 0, stream>>>(X16, 256, 0, WVT, 0, bv, 0,
      nullptr, 0, 0, V0b, VIN_, 0, VIN_, 256);
  wmma_gemm_kernel<true,false><<<dim3(64, 1, U_), 256, 0, stream>>>(HS16, (long)U_*H_, H_, WQT, (long)128*512, bq, 64,
      nullptr, 0, 0, QBb, (long)U_*64, 64, 64, 512);

  // 4) routing + act assembly
  router_kernel<<<B_, 128, 0, stream>>>(KINb, QBb, V0b, bk, bv, HS16, ACTb, MASKb);

  // 5) gates GEMM  [B,928] x [928,2048] per u
  wmma_gemm_kernel<false,false><<<dim3(64, 16, U_), 256, 0, stream>>>(ACTb, (long)U_*KGATE, KGATE, WCAT, (long)2048*KGATE,
      nullptr, 0, nullptr, 0, 0, GATESb, (long)U_*2048, 2048, 2048, KGATE);

  // 6) LSTM pointwise -> c_new (d_out), h_new (f32+f16)
  lstm_kernel<<<(B_*U_*H_ + 255)/256, 256, 0, stream>>>(GATESb, cs, bih, bhh, cnew, HNEWb, HNEW16, (long)B_*U_*H_);

  // 7) qc|kc|vc GEMM  [B,512] x [512,2304] per u
  wmma_gemm_kernel<false,false><<<dim3(64, 18, U_), 256, 0, stream>>>(HNEW16, (long)U_*H_, H_, WQKVT, (long)NQKV*512,
      nullptr, 0, nullptr, 0, 0, QKVb, (long)U_*NQKV, NQKV, NQKV, 512);

  // 8) per-(b,head) attention -> ctx f16
  attn_kernel<<<dim3(B_, HC_), 256, 0, stream>>>(QKVb, bqc, bkc, bvc, MASKb, CTX16b);

  // 9) output GEMM [B,2048] x [2048,512] per u, epilogue += bo + h_new -> hs_final
  wmma_gemm_kernel<true,true><<<dim3(64, 4, U_), 256, 0, stream>>>(CTX16b, (long)U_*HC_*VC_, (long)HC_*VC_, WOT, (long)512*2048,
      bo, 512, HNEWb, (long)U_*H_, H_, hsfin, (long)U_*H_, H_, 512, 2048);
}
